// BVPGenerator_15152644621000
// MI455X (gfx1250) — matrix operand order repaired and checked by reading, not compiled, not run on hardware
//
#include <hip/hip_runtime.h>
#include <hip/hip_bf16.h>
#include <math.h>

// ---------------------------------------------------------------------------
// Problem constants
#define LAT   128
#define HIDN  256
#define TSTEP 1920
#define BATCH 128
#define H2    128

typedef float  v8f   __attribute__((ext_vector_type(8)));
typedef __bf16 v16bf __attribute__((ext_vector_type(16)));
typedef unsigned int u32x4 __attribute__((ext_vector_type(4)));

union AFrag { u32x4 q[2]; v16bf v; };

__device__ __forceinline__ float sigm(float x) { return 1.0f / (1.0f + __expf(-x)); }

__device__ __forceinline__ float fast_tanh(float x) {
#if __has_builtin(__builtin_amdgcn_tanhf)
    return __builtin_amdgcn_tanhf(x);        // V_TANH_F32 (CDNA5 trans op)
#elif __has_builtin(__builtin_amdgcn_tanh_f32)
    return __builtin_amdgcn_tanh_f32(x);
#else
    float e = __expf(2.0f * x);
    return (e - 1.0f) / (e + 1.0f);
#endif
}

// Opaque zero offset (SGPR): redefined by empty asm each call so loads whose
// address depends on it cannot be hoisted/CSE'd across loop iterations, while
// the base pointer itself stays a kernarg -> address-space inference keeps the
// loads in the GLOBAL path (global_load_b128, LOADcnt only), not FLAT.
__device__ __forceinline__ unsigned opaque_zero() {
    unsigned v = 0;
    asm volatile("" : "+s"(v));
    return v;
}

// B fragment (32x16 bf16, K x N): lane L holds column n = L&15,
// K = kt*32 + (L>>4)*16 .. +16  -> 32 contiguous bytes of W[n_col][k] (row-major (col,k))
__device__ __forceinline__ v16bf load_b_frag(const __bf16* W, int col, int kt, int lane) {
    const __bf16* p = W + (size_t)col * HIDN + kt * 32 + ((lane >> 4) << 4);
    return *(const v16bf*)p;
}

// A fragment (16x32 bf16, M x K) from a row-major (M, HIDN) buffer:
// lane L: M = L&15, Khalf = L>>4; values = K run [kt*32+16*Khalf, +16).
__device__ __forceinline__ v16bf load_a_frag(const __bf16* h, long rowstride_row0, int kt, int lane) {
    int m = lane & 15, kh = lane >> 4;
    const __bf16* p = h + (rowstride_row0 + (long)m * HIDN) + kt * 32 + kh * 16;
    AFrag a;
    a.q[0] = *(const u32x4*)p;
    a.q[1] = *(const u32x4*)(p + 8);
    return a.v;
}

__device__ __forceinline__ v8f wmma_bf16(v16bf A, v16bf B, v8f C) {
    return __builtin_amdgcn_wmma_f32_16x16x32_bf16(false, A, false, B, (short)0, C, false, false);
}

// ---------------------------------------------------------------------------
// f32 -> bf16 weight conversion
__global__ void convert_kernel(const float* __restrict__ src, __bf16* __restrict__ dst, int n) {
    int i = blockIdx.x * blockDim.x + threadIdx.x;
    if (i < n) dst[i] = (__bf16)src[i];
}

// ---------------------------------------------------------------------------
// Precursor network + one-time layer-0 input GEMM.
// One block per batch row, 256 threads.
__global__ void __launch_bounds__(256) prep_kernel(
    const float* __restrict__ z, const int* __restrict__ labels,
    const float* __restrict__ emb_W, const float* __restrict__ np_W,
    const float* __restrict__ np_b, const float* __restrict__ np_g,
    const float* __restrict__ np_be, const float* __restrict__ Wih0,
    const float* __restrict__ bih0, const float* __restrict__ bhh0,
    float* __restrict__ xw0)
{
    __shared__ float embs[HIDN];
    __shared__ float cat[2 * HIDN];
    __shared__ float red[256];
    __shared__ float red2[256];
    int t = threadIdx.x, b = blockIdx.x;
    int lab = labels[b];
    embs[t] = emb_W[lab * HIDN + t];
    __syncthreads();

    const float* zr = z + b * LAT;
    const float* wr = np_W + (size_t)t * (LAT + HIDN);
    float acc = np_b[t];
    #pragma unroll 4
    for (int k = 0; k < LAT; ++k) acc += zr[k] * wr[k];
    #pragma unroll 4
    for (int k = 0; k < HIDN; ++k) acc += embs[k] * wr[LAT + k];
    red[t] = acc; red2[t] = acc * acc;
    __syncthreads();
    for (int s = 128; s > 0; s >>= 1) {
        if (t < s) { red[t] += red[t + s]; red2[t] += red2[t + s]; }
        __syncthreads();
    }
    float mean = red[0] * (1.0f / HIDN);
    float var  = red2[0] * (1.0f / HIDN) - mean * mean;
    float y = (acc - mean) * rsqrtf(var + 1e-5f) * np_g[t] + np_be[t];
    y = y > 0.0f ? y : 0.2f * y;
    cat[t] = y;
    cat[HIDN + t] = embs[t];
    __syncthreads();

    for (int g = t; g < 4 * HIDN; g += 256) {
        const float* wi = Wih0 + (size_t)g * (2 * HIDN);
        float v = bih0[g] + bhh0[g];
        #pragma unroll 4
        for (int k = 0; k < 2 * HIDN; ++k) v += cat[k] * wi[k];
        xw0[(size_t)b * 4 * HIDN + g] = v;
    }
}

// ---------------------------------------------------------------------------
// Persistent two-layer LSTM. 8 blocks (16 batch rows each) x 512 threads
// (16 waves; wave w owns hidden units [16w, 16w+16) -> i/f/g/o gate columns).
// h0/h1 in LDS (bf16), c0/c1 in VGPRs, weights streamed from L2 as bf16
// every step (opaque offset blocks LICM; GLOBAL path keeps DScnt free for
// the LDS A-fragment loads).
__global__ void __launch_bounds__(512) lstm_persist(
    const float* __restrict__ xw0,
    const __bf16* __restrict__ Whh0b, const __bf16* __restrict__ Wih1b,
    const __bf16* __restrict__ Whh1b,
    const float* __restrict__ bih1, const float* __restrict__ bhh1,
    __bf16* __restrict__ ch, float* __restrict__ hsum)
{
    __shared__ __bf16 h0s[16 * HIDN];
    __shared__ __bf16 h1s[16 * HIDN];

    int tid  = threadIdx.x;
    int lane = tid & 31;
    int w    = tid >> 5;          // wave id 0..15
    int b0   = blockIdx.x * 16;   // global batch base for this block
    int j0   = w * 16;            // hidden units owned by this wave
    int n    = lane & 15;         // N within tile (hidden unit / gate col)
    int kh   = lane >> 4;         // lane half -> M offset 0/8 in acc layout

    for (int i = tid; i < 16 * HIDN; i += 512) {
        h0s[i] = (__bf16)0.0f;
        h1s[i] = (__bf16)0.0f;
    }
    __syncthreads();

    const int gi = j0, gf = HIDN + j0, gg = 2 * HIDN + j0, go = 3 * HIDN + j0;

    // Time-invariant layer-0 gate pre-activations (acc layout:
    // component r of lane -> batch m = r + 8*kh, column = gate_base + n).
    float xi[8], xf[8], xg[8], xo[8];
    #pragma unroll
    for (int r = 0; r < 8; ++r) {
        const float* row = xw0 + (size_t)(b0 + r + 8 * kh) * (4 * HIDN);
        xi[r] = row[gi + n]; xf[r] = row[gf + n];
        xg[r] = row[gg + n]; xo[r] = row[go + n];
    }
    const float b1i = bih1[gi + n] + bhh1[gi + n];
    const float b1f = bih1[gf + n] + bhh1[gf + n];
    const float b1g = bih1[gg + n] + bhh1[gg + n];
    const float b1o = bih1[go + n] + bhh1[go + n];

    float c0[8] = {}, c1[8] = {}, hs[8] = {};

    for (int t = 0; t < TSTEP; ++t) {
        // Per-iteration opaque zero offset: weight-fragment loads stay in the
        // loop (streamed from the L2-resident bf16 copies) without spilling.
        unsigned zoff = opaque_zero();
        const __bf16* W0  = Whh0b + zoff;
        const __bf16* W1i = Wih1b + zoff;
        const __bf16* W1h = Whh1b + zoff;

        // ---------------- layer 0: gates = xw0 + Whh0 . h0_{t-1} ----------
        v8f ai, af, ag, ao;
        #pragma unroll
        for (int r = 0; r < 8; ++r) { ai[r] = xi[r]; af[r] = xf[r]; ag[r] = xg[r]; ao[r] = xo[r]; }
        #pragma unroll
        for (int kt = 0; kt < 8; ++kt) {
            v16bf A  = load_a_frag(h0s, 0, kt, lane);
            v16bf Bi = load_b_frag(W0, gi + n, kt, lane);
            v16bf Bf = load_b_frag(W0, gf + n, kt, lane);
            v16bf Bg = load_b_frag(W0, gg + n, kt, lane);
            v16bf Bo = load_b_frag(W0, go + n, kt, lane);
            ai = wmma_bf16(A, Bi, ai);
            af = wmma_bf16(A, Bf, af);
            ag = wmma_bf16(A, Bg, ag);
            ao = wmma_bf16(A, Bo, ao);
        }
        __syncthreads();  // everyone done reading h0_{t-1}
        float h0new[8];
        #pragma unroll
        for (int r = 0; r < 8; ++r) {
            float I = sigm(ai[r]), F = sigm(af[r]);
            float G = fast_tanh(ag[r]), O = sigm(ao[r]);
            c0[r] = F * c0[r] + I * G;
            h0new[r] = O * fast_tanh(c0[r]);
        }
        #pragma unroll
        for (int r = 0; r < 8; ++r)
            h0s[(r + 8 * kh) * HIDN + j0 + n] = (__bf16)h0new[r];
        __syncthreads();  // h0s now h0_t

        // ------- layer 1: gates = b1 + Wih1 . h0_t + Whh1 . h1_{t-1} ------
        #pragma unroll
        for (int r = 0; r < 8; ++r) { ai[r] = b1i; af[r] = b1f; ag[r] = b1g; ao[r] = b1o; }
        #pragma unroll
        for (int kt = 0; kt < 8; ++kt) {
            v16bf A0 = load_a_frag(h0s, 0, kt, lane);   // current h0
            v16bf A1 = load_a_frag(h1s, 0, kt, lane);   // previous h1
            ai = wmma_bf16(A0, load_b_frag(W1i, gi + n, kt, lane), ai);
            ai = wmma_bf16(A1, load_b_frag(W1h, gi + n, kt, lane), ai);
            af = wmma_bf16(A0, load_b_frag(W1i, gf + n, kt, lane), af);
            af = wmma_bf16(A1, load_b_frag(W1h, gf + n, kt, lane), af);
            ag = wmma_bf16(A0, load_b_frag(W1i, gg + n, kt, lane), ag);
            ag = wmma_bf16(A1, load_b_frag(W1h, gg + n, kt, lane), ag);
            ao = wmma_bf16(A0, load_b_frag(W1i, go + n, kt, lane), ao);
            ao = wmma_bf16(A1, load_b_frag(W1h, go + n, kt, lane), ao);
        }
        __syncthreads();  // everyone done reading h1_{t-1}
        #pragma unroll
        for (int r = 0; r < 8; ++r) {
            float I = sigm(ai[r]), F = sigm(af[r]);
            float G = fast_tanh(ag[r]), O = sigm(ao[r]);
            c1[r] = F * c1[r] + I * G;
            float h = O * fast_tanh(c1[r]);
            hs[r] += h;
            int mb = r + 8 * kh;
            h1s[mb * HIDN + j0 + n] = (__bf16)h;
            // Streamed once, consumed once by base_head -> non-temporal.
            __builtin_nontemporal_store(
                (__bf16)h, ch + ((size_t)(b0 + mb) * TSTEP + t) * HIDN + j0 + n);
        }
        __syncthreads();  // h1s now h1_t
    }

    #pragma unroll
    for (int r = 0; r < 8; ++r)
        hsum[(size_t)(b0 + r + 8 * kh) * HIDN + j0 + n] = hs[r];
}

// ---------------------------------------------------------------------------
// Base (sample) head: for each of B*T rows, tanh(leaky(LN(ch@sW1.T+sb1))@sW2.T+sb2).
// 1920 blocks x 256 threads; each wave owns one 16-row M-tile, full N=128.
__global__ void __launch_bounds__(256) base_head(
    const __bf16* __restrict__ ch, const __bf16* __restrict__ sW1b,
    const float* __restrict__ sb1, const float* __restrict__ sg,
    const float* __restrict__ sbe, const float* __restrict__ sW2,
    const float* __restrict__ sb2, float* __restrict__ base)
{
    int tid  = threadIdx.x;
    int lane = tid & 31;
    int w    = tid >> 5;
    int n    = lane & 15;
    int kh   = lane >> 4;
    long row0 = (long)blockIdx.x * 128 + (long)w * 16;

    float sgn[8], sben[8], sw2n[8];
    v8f acc[8];
    #pragma unroll
    for (int nt = 0; nt < 8; ++nt) {
        int col = nt * 16 + n;
        float sb = sb1[col];
        sgn[nt] = sg[col]; sben[nt] = sbe[col]; sw2n[nt] = sW2[col];
        #pragma unroll
        for (int r = 0; r < 8; ++r) acc[nt][r] = sb;
    }
    #pragma unroll
    for (int kt = 0; kt < 8; ++kt) {
        v16bf A = load_a_frag(ch, row0 * HIDN, kt, lane);
        #pragma unroll
        for (int nt = 0; nt < 8; ++nt)
            acc[nt] = wmma_bf16(A, load_b_frag(sW1b, nt * 16 + n, kt, lane), acc[nt]);
    }

    float sb2v = sb2[0];
    #pragma unroll
    for (int r = 0; r < 8; ++r) {
        float s1 = 0.0f, s2 = 0.0f;
        #pragma unroll
        for (int nt = 0; nt < 8; ++nt) { float y = acc[nt][r]; s1 += y; s2 += y * y; }
        #pragma unroll
        for (int m = 1; m < 16; m <<= 1) {
            s1 += __shfl_xor(s1, m, 32);
            s2 += __shfl_xor(s2, m, 32);
        }
        float mean = s1 * (1.0f / H2);
        float var  = s2 * (1.0f / H2) - mean * mean;
        float rstd = rsqrtf(var + 1e-5f);
        float dot = 0.0f;
        #pragma unroll
        for (int nt = 0; nt < 8; ++nt) {
            float y = (acc[nt][r] - mean) * rstd * sgn[nt] + sben[nt];
            y = y > 0.0f ? y : 0.2f * y;
            dot += y * sw2n[nt];
        }
        #pragma unroll
        for (int m = 1; m < 16; m <<= 1) dot += __shfl_xor(dot, m, 32);
        if (n == 0) base[row0 + r + 8 * kh] = fast_tanh(dot + sb2v);
    }
}

// ---------------------------------------------------------------------------
// Oscillator-parameter head: op = leaky(LN(h_avg@oW1.T+ob1))@oW2.T+ob2.
// One block per batch row, 128 threads.
__global__ void __launch_bounds__(128) op_head(
    const float* __restrict__ hsum, const float* __restrict__ oW1,
    const float* __restrict__ ob1, const float* __restrict__ og,
    const float* __restrict__ obe, const float* __restrict__ oW2,
    const float* __restrict__ ob2, float* __restrict__ prm)
{
    __shared__ float havg[HIDN];
    __shared__ float red[128], red2[128];
    int t = threadIdx.x, b = blockIdx.x;
    havg[t]       = hsum[b * HIDN + t]       * (1.0f / TSTEP);
    havg[t + 128] = hsum[b * HIDN + t + 128] * (1.0f / TSTEP);
    __syncthreads();
    float acc = ob1[t];
    const float* wr = oW1 + (size_t)t * HIDN;
    #pragma unroll 4
    for (int k = 0; k < HIDN; ++k) acc += havg[k] * wr[k];
    red[t] = acc; red2[t] = acc * acc;
    __syncthreads();
    for (int s = 64; s > 0; s >>= 1) {
        if (t < s) { red[t] += red[t + s]; red2[t] += red2[t + s]; }
        __syncthreads();
    }
    float mean = red[0] * (1.0f / H2);
    float var  = red2[0] * (1.0f / H2) - mean * mean;
    float y = (acc - mean) * rsqrtf(var + 1e-5f) * og[t] + obe[t];
    y = y > 0.0f ? y : 0.2f * y;
    float op[3];
    for (int c = 0; c < 3; ++c) {
        __syncthreads();
        red[t] = y * oW2[c * H2 + t];
        __syncthreads();
        for (int s = 64; s > 0; s >>= 1) {
            if (t < s) red[t] += red[t + s];
            __syncthreads();
        }
        op[c] = red[0] + ob2[c];
    }
    if (t == 0) {
        prm[b * 3 + 0] = 0.23f + 0.04f * fast_tanh(op[0]);
        prm[b * 3 + 1] = 2.0f + 1.5f * fast_tanh(op[1]);
        prm[b * 3 + 2] = 3.14159265358979323f * sigm(op[2]);
    }
}

// ---------------------------------------------------------------------------
// Final combine: osc + enhanced/stressed/smoothed + label select.
__global__ void finalize_k(
    const float* __restrict__ base, const float* __restrict__ prm,
    const int* __restrict__ labels, const float* __restrict__ stress_w,
    const float* __restrict__ amu_w, const float* __restrict__ amu_b,
    float* __restrict__ out)
{
    int idx = blockIdx.x * blockDim.x + threadIdx.x;
    if (idx >= BATCH * TSTEP) return;
    int b = idx / TSTEP, t = idx % TSTEP;
    float freq = prm[b * 3 + 0], amp = prm[b * 3 + 1], ph = prm[b * 3 + 2];
    float w0 = 6.283185307179586f * freq * (float)TSTEP / (float)(TSTEP - 1);
    const float* bb = base + (size_t)b * TSTEP;
    auto enh = [&](int tt) -> float {
        if (tt < 0 || tt >= TSTEP) return 0.0f;
        float osc = amp * sinf(w0 * (float)tt + ph);
        return 0.6f * bb[tt] + 0.4f * osc;
    };
    float e = enh(t);
    int lab = labels[b];
    float r;
    if (lab == 1)      r = e;
    else if (lab == 2) r = stress_w[0] * e;
    else if (lab == 3) r = amu_w[0] * enh(t - 1) + amu_w[1] * e + amu_w[2] * enh(t + 1) + amu_b[0];
    else               r = 0.0f;
    out[idx] = r;
}

// ---------------------------------------------------------------------------
extern "C" void kernel_launch(void* const* d_in, const int* in_sizes, int n_in,
                              void* d_out, int out_size, void* d_ws, size_t ws_size,
                              hipStream_t stream) {
    const float* z      = (const float*)d_in[0];
    const int*   labels = (const int*)  d_in[1];
    const float* emb_W  = (const float*)d_in[2];
    const float* np_W   = (const float*)d_in[3];
    const float* np_b   = (const float*)d_in[4];
    const float* np_g   = (const float*)d_in[5];
    const float* np_be  = (const float*)d_in[6];
    const float* Wih0   = (const float*)d_in[7];
    const float* Whh0   = (const float*)d_in[8];
    const float* bih0   = (const float*)d_in[9];
    const float* bhh0   = (const float*)d_in[10];
    const float* Wih1   = (const float*)d_in[11];
    const float* Whh1   = (const float*)d_in[12];
    const float* bih1   = (const float*)d_in[13];
    const float* bhh1   = (const float*)d_in[14];
    const float* oW1    = (const float*)d_in[15];
    const float* ob1    = (const float*)d_in[16];
    const float* og     = (const float*)d_in[17];
    const float* obe    = (const float*)d_in[18];
    const float* oW2    = (const float*)d_in[19];
    const float* ob2    = (const float*)d_in[20];
    const float* sW1    = (const float*)d_in[21];
    const float* sb1    = (const float*)d_in[22];
    const float* sg     = (const float*)d_in[23];
    const float* sbe    = (const float*)d_in[24];
    const float* sW2    = (const float*)d_in[25];
    const float* sb2    = (const float*)d_in[26];
    const float* stress = (const float*)d_in[27];
    const float* amu_w  = (const float*)d_in[28];
    const float* amu_b  = (const float*)d_in[29];

    // Workspace layout (bytes, all offsets 256-aligned)
    char* ws = (char*)d_ws;
    float*  xw0   = (float*) (ws + 0);          // 128*1024 f32     = 524288
    __bf16* Whh0b = (__bf16*)(ws + 524288);     // 1024*256 bf16    = 524288
    __bf16* Wih1b = (__bf16*)(ws + 1048576);    // 1024*256 bf16    = 524288
    __bf16* Whh1b = (__bf16*)(ws + 1572864);    // 1024*256 bf16    = 524288
    __bf16* sW1b  = (__bf16*)(ws + 2097152);    // 128*256 bf16     = 65536
    float*  hsum  = (float*) (ws + 2162688);    // 128*256 f32      = 131072
    float*  prm   = (float*) (ws + 2293760);    // 128*3 f32        = 1536 (pad to 256)
    float*  base  = (float*) (ws + 2295296);    // 128*1920 f32     = 983040
    __bf16* ch    = (__bf16*)(ws + 3278336);    // 128*1920*256 bf16 = 125829120

    // 1) weight conversion f32 -> bf16
    int nw = 1024 * 256;
    convert_kernel<<<(nw + 255) / 256, 256, 0, stream>>>(Whh0, Whh0b, nw);
    convert_kernel<<<(nw + 255) / 256, 256, 0, stream>>>(Wih1, Wih1b, nw);
    convert_kernel<<<(nw + 255) / 256, 256, 0, stream>>>(Whh1, Whh1b, nw);
    convert_kernel<<<(128 * 256 + 255) / 256, 256, 0, stream>>>(sW1, sW1b, 128 * 256);

    // 2) precursor network + one-time layer-0 input GEMM
    prep_kernel<<<BATCH, 256, 0, stream>>>(z, labels, emb_W, np_W, np_b, np_g,
                                           np_be, Wih0, bih0, bhh0, xw0);

    // 3) persistent two-layer WMMA LSTM (batch-parallel over 8 WGPs)
    lstm_persist<<<8, 512, 0, stream>>>(xw0, Whh0b, Wih1b, Whh1b, bih1, bhh1, ch, hsum);

    // 4) GPU-wide WMMA base head over B*T rows
    base_head<<<(BATCH * TSTEP) / 128, 256, 0, stream>>>(ch, sW1b, sb1, sg, sbe,
                                                         sW2, sb2, base);

    // 5) oscillator-parameter head
    op_head<<<BATCH, 128, 0, stream>>>(hsum, oW1, ob1, og, obe, oW2, ob2, prm);

    // 6) final combine + label select
    finalize_k<<<(BATCH * TSTEP) / 256, 256, 0, stream>>>(base, prm, labels, stress,
                                                          amu_w, amu_b, (float*)d_out);
}
